// Decoder_14267881357775
// MI455X (gfx1250) — matrix operand order repaired and checked by reading, not compiled, not run on hardware
//
#include <hip/hip_runtime.h>
#include <hip/hip_bf16.h>

// ---------------------------------------------------------------------------
// Deep-Voice-3 style decoder on MI455X (gfx1250, wave32).
// All matmuls run on v_wmma_f32_16x16x32_bf16 (fp32 -> bf16 inputs, f32 acc).
// - Weights converted + pre-swizzled into WMMA B-fragment order once per
//   launch (lane's 16 elems contiguous -> 2x global_load_b128, zero pack VALU)
// - Each GEMM wave computes a 32x32 C tile: 2 A frags x 2 B frags = 4 WMMAs
// - v_out is produced transposed so the K=512 ctx GEMM gets k-contiguous
//   vectorized B loads (4x b128 + 8x v_cvt_pk_bf16_f32 per fragment)
// Workspace requirement: ~210 MB.
// ---------------------------------------------------------------------------

typedef __attribute__((ext_vector_type(16))) __bf16 v16bf;
typedef __attribute__((ext_vector_type(8)))  float  v8f;

#define SQRT_HALF 0.70710678118654752f
#define LN10000   9.21034037197618274f

__device__ __forceinline__ __bf16 f2bf(float f) { return (__bf16)f; }  // native cvt

__device__ __forceinline__ v8f wmma_bf16(v16bf a, v16bf b, v8f c) {
    return __builtin_amdgcn_wmma_f32_16x16x32_bf16(false, a, false, b, (short)0, c,
                                                   false, false);
}

// pos*pos*rate / 10000^(2*(c//2)/256) ; even c -> sin, odd c -> cos
__device__ __forceinline__ float posenc(int t, int c, float rate) {
    float inv = __expf(-LN10000 * (float)(c & ~1) * (1.0f / 256.0f));
    float ang = (float)t * ((float)t * rate * inv);
    return (c & 1) ? __cosf(ang) : __sinf(ang);
}

// ---- fragment loaders (ISA 7.12.2 layouts, wave32) -------------------------

// A 16x32 bf16 from fp32: lanes 0-15 = rows; contiguous 8-float runs per half
// vectorize to global_load_b128; converts via v_cvt_pk_bf16_f32.
template <bool POSENC>
__device__ __forceinline__ v16bf load_a_frag(const float* __restrict__ A, int lda,
                                             int m0, int k0, int lane, float rate) {
    int row  = m0 + (lane & 15);
    int half = lane >> 4;
    const float* ap = A + (size_t)row * lda + k0;
    v16bf a;
#pragma unroll
    for (int v = 0; v < 8; ++v) {
        int kb = (v < 4) ? (half * 8 + 2 * v) : (16 + half * 8 + 2 * (v - 4));
#pragma unroll
        for (int e = 0; e < 2; ++e) {
            float x = ap[kb + e];
            if (POSENC) x += posenc(row, k0 + kb + e, rate);
            a[2 * v + e] = f2bf(x);
        }
    }
    return a;
}

// Pre-swizzled bf16 weight: 32x16 K-tile x N-tile grid, 512 elems (1 KB) per
// tile, each lane's 16 elements contiguous (32 B) -> 2x b128 loads.
__device__ __forceinline__ v16bf load_b_frag_swz(const __bf16* __restrict__ B,
                                                 int tilesN, int k0, int n0, int lane) {
    const __bf16* p = B + ((size_t)(k0 >> 5) * tilesN + (n0 >> 4)) * 512 + lane * 16;
    return *(const v16bf*)p;
}

// B(k,n) = Bsrc[n*ldb + k]  (fp32 activation with k contiguous per lane:
// k_out [Tk,C] for q@k^T, and transposed v_out [C,Tk] for probs@v)
__device__ __forceinline__ v16bf load_b_frag_f32t(const float* __restrict__ B, int ldb,
                                                  int k0, int n0, int lane) {
    int n  = n0 + (lane & 15);
    int kb = k0 + ((lane >> 4) << 3);
    const float* bp = B + (size_t)n * ldb + kb;
    v16bf b;
#pragma unroll
    for (int i = 0; i < 16; ++i) b[i] = f2bf(bp[i < 8 ? i : i + 8]);
    return b;
}

// ---- generic batched GEMM: C[b] = epi(alpha * A[b]@B + bias) ---------------
// Each wave computes a 32x32 tile: 2 A fragments x 2 B fragments -> 4 WMMAs.
// EPI: 0 = none, 1 = relu, 2 = (.. + resid) * sqrt(0.5)
// BMODE: 0 = swizzled bf16 weight (ldb = N>>4 tile pitch, batch-invariant),
//        1 = fp32 per-batch, k-contiguous (B(k,n) = Bsrc[n*ldb+k])
// STORE_T: store C transposed: C[n*ldc + m] (m-contiguous stores)
template <int EPI, int BMODE, bool POSENC, bool STORE_T>
__global__ void gemm_wmma(const float* __restrict__ A, long long sA, int lda,
                          const void* __restrict__ B, long long sB, int ldb,
                          const float* __restrict__ bias,
                          const float* __restrict__ resid, long long sR,
                          float* __restrict__ C, long long sC, int ldc,
                          int M, int N, int K, float alpha, float rate) {
    int lane   = threadIdx.x & 31;
    int wave   = threadIdx.x >> 5;
    int tilesN = N >> 5;                           // 32-wide tiles
    int tile   = blockIdx.x * (blockDim.x >> 5) + wave;
    if (tile >= (M >> 5) * tilesN) return;         // wave-uniform: EXEC stays full
    int b  = blockIdx.y;
    int m0 = (tile / tilesN) << 5;
    int n0 = (tile % tilesN) << 5;

    const float* Ab = A + (size_t)b * sA;
    v8f acc00 = {}, acc01 = {}, acc10 = {}, acc11 = {};
    for (int k0 = 0; k0 < K; k0 += 32) {
        v16bf a0 = load_a_frag<POSENC>(Ab, lda, m0, k0, lane, rate);
        v16bf a1 = load_a_frag<POSENC>(Ab, lda, m0 + 16, k0, lane, rate);
        v16bf b0, b1;
        if constexpr (BMODE == 0) {
            b0 = load_b_frag_swz((const __bf16*)B, ldb, k0, n0, lane);
            b1 = load_b_frag_swz((const __bf16*)B, ldb, k0, n0 + 16, lane);
        } else {
            const float* Bb = (const float*)B + (size_t)b * sB;
            b0 = load_b_frag_f32t(Bb, ldb, k0, n0, lane);
            b1 = load_b_frag_f32t(Bb, ldb, k0, n0 + 16, lane);
        }
        acc00 = wmma_bf16(a0, b0, acc00);
        acc01 = wmma_bf16(a0, b1, acc01);
        acc10 = wmma_bf16(a1, b0, acc10);
        acc11 = wmma_bf16(a1, b1, acc11);
    }

    int n  = n0 + (lane & 15);
    int mb = (lane >> 4) << 3;
    float bv0 = bias ? bias[n] : 0.0f;
    float bv1 = bias ? bias[n + 16] : 0.0f;
    float* Cb = C + (size_t)b * sC;
    const float* Rb = resid ? (resid + (size_t)b * sR) : nullptr;
#pragma unroll
    for (int r = 0; r < 8; ++r) {
        int ma = m0 + mb + r;
        int mc = m0 + 16 + mb + r;
        float v00 = acc00[r] * alpha + bv0;
        float v01 = acc01[r] * alpha + bv1;
        float v10 = acc10[r] * alpha + bv0;
        float v11 = acc11[r] * alpha + bv1;
        if (EPI == 1) {
            v00 = fmaxf(v00, 0.0f); v01 = fmaxf(v01, 0.0f);
            v10 = fmaxf(v10, 0.0f); v11 = fmaxf(v11, 0.0f);
        }
        if (EPI == 2) {
            v00 = (v00 + Rb[(size_t)ma * ldc + n])      * SQRT_HALF;
            v01 = (v01 + Rb[(size_t)ma * ldc + n + 16]) * SQRT_HALF;
            v10 = (v10 + Rb[(size_t)mc * ldc + n])      * SQRT_HALF;
            v11 = (v11 + Rb[(size_t)mc * ldc + n + 16]) * SQRT_HALF;
        }
        if constexpr (STORE_T) {                    // C[n*ldc + m], m-contiguous
            Cb[(size_t)n * ldc + ma]        = v00;
            Cb[(size_t)(n + 16) * ldc + ma] = v01;
            Cb[(size_t)n * ldc + mc]        = v10;
            Cb[(size_t)(n + 16) * ldc + mc] = v11;
        } else {
            Cb[(size_t)ma * ldc + n]      = v00;
            Cb[(size_t)ma * ldc + n + 16] = v01;
            Cb[(size_t)mc * ldc + n]      = v10;
            Cb[(size_t)mc * ldc + n + 16] = v11;
        }
    }
}

// ---- causal dilated conv (K=5) + GLU + residual ----------------------------
// Per wave: 32 rows x paired (a,g) 16-col tiles -> 4 WMMAs per k-step.
// X: [B,T,256] fp32, W: swizzled bf16 (K=1280 x N=512 fragment grid)
__global__ void conv_glu_wmma(const float* __restrict__ X, const __bf16* __restrict__ W,
                              const float* __restrict__ bias, float* __restrict__ Y,
                              int T, int dil) {
    int lane = threadIdx.x & 31;
    int wave = threadIdx.x >> 5;
    int tile = blockIdx.x * (blockDim.x >> 5) + wave;
    if (tile >= (T >> 5) * 16) return;
    int b  = blockIdx.y;
    int m0 = (tile / 16) << 5;
    int n0 = (tile % 16) << 4;
    const float* Xb = X + (size_t)b * T * 256;

    v8f acc_a0 = {}, acc_g0 = {}, acc_a1 = {}, acc_g1 = {};
    for (int j = 0; j < 5; ++j) {
        int shift = (j - 4) * dil;                 // w[4] taps the current sample
        for (int k0 = 0; k0 < 256; k0 += 32) {
            int half = lane >> 4;
            v16bf a0, a1;
#pragma unroll
            for (int i = 0; i < 16; ++i) { a0[i] = f2bf(0.0f); a1[i] = f2bf(0.0f); }
            int row0 = m0 + (lane & 15) + shift;
            int row1 = row0 + 16;
            if (row0 >= 0) {
                const float* ap = Xb + (size_t)row0 * 256 + k0;
#pragma unroll
                for (int v = 0; v < 8; ++v) {
                    int kb = (v < 4) ? (half * 8 + 2 * v) : (16 + half * 8 + 2 * (v - 4));
                    a0[2 * v]     = f2bf(ap[kb]);
                    a0[2 * v + 1] = f2bf(ap[kb + 1]);
                }
            }
            if (row1 >= 0) {
                const float* ap = Xb + (size_t)row1 * 256 + k0;
#pragma unroll
                for (int v = 0; v < 8; ++v) {
                    int kb = (v < 4) ? (half * 8 + 2 * v) : (16 + half * 8 + 2 * (v - 4));
                    a1[2 * v]     = f2bf(ap[kb]);
                    a1[2 * v + 1] = f2bf(ap[kb + 1]);
                }
            }
            v16bf ba = load_b_frag_swz(W, 32, j * 256 + k0, n0, lane);
            v16bf bg = load_b_frag_swz(W, 32, j * 256 + k0, n0 + 256, lane);
            acc_a0 = wmma_bf16(a0, ba, acc_a0);
            acc_g0 = wmma_bf16(a0, bg, acc_g0);
            acc_a1 = wmma_bf16(a1, ba, acc_a1);
            acc_g1 = wmma_bf16(a1, bg, acc_g1);
        }
    }

    int n  = n0 + (lane & 15);
    int mb = (lane >> 4) << 3;
    float bav = bias[n], bgv = bias[n + 256];
    float* Yb = Y + (size_t)b * T * 256;
#pragma unroll
    for (int r = 0; r < 8; ++r) {
        int m0r = m0 + mb + r;
        int m1r = m0 + 16 + mb + r;
        float aa0 = acc_a0[r] + bav, gg0 = acc_g0[r] + bgv;
        float aa1 = acc_a1[r] + bav, gg1 = acc_g1[r] + bgv;
        float s0 = 1.0f / (1.0f + __expf(-gg0));
        float s1 = 1.0f / (1.0f + __expf(-gg1));
        Yb[(size_t)m0r * 256 + n] = (Xb[(size_t)m0r * 256 + n] + aa0 * s0) * SQRT_HALF;
        Yb[(size_t)m1r * 256 + n] = (Xb[(size_t)m1r * 256 + n] + aa1 * s1) * SQRT_HALF;
    }
}

// ---- softmax over rows of length 512: one wave32 per row -------------------
__global__ void softmax512(float* __restrict__ AW, int rows) {
    int lane = threadIdx.x & 31;
    int wave = threadIdx.x >> 5;
    int row  = blockIdx.x * (blockDim.x >> 5) + wave;
    if (row >= rows) return;
    float* p = AW + (size_t)row * 512;
    float vals[16];
    float mx = -3.4e38f;
#pragma unroll
    for (int i = 0; i < 16; ++i) { vals[i] = p[lane + 32 * i]; mx = fmaxf(mx, vals[i]); }
#pragma unroll
    for (int off = 16; off; off >>= 1) mx = fmaxf(mx, __shfl_xor(mx, off, 32));
    float sum = 0.0f;
#pragma unroll
    for (int i = 0; i < 16; ++i) { vals[i] = __expf(vals[i] - mx); sum += vals[i]; }
#pragma unroll
    for (int off = 16; off; off >>= 1) sum += __shfl_xor(sum, off, 32);
    float inv = 1.0f / sum;
#pragma unroll
    for (int i = 0; i < 16; ++i) p[lane + 32 * i] = vals[i] * inv;
}

// ---- done head: sigmoid(h @ w_done + b_done), N=2 (too thin for WMMA) ------
__global__ void done_head(const float* __restrict__ H, const float* __restrict__ W,
                          const float* __restrict__ B, float* __restrict__ O, int rows) {
    int r = blockIdx.x * blockDim.x + threadIdx.x;
    if (r >= rows) return;
    const float* h = H + (size_t)r * 256;
    float s0 = B[0], s1 = B[1];
    for (int c = 0; c < 256; ++c) {
        float hv = h[c];
        s0 += hv * W[c * 2];
        s1 += hv * W[c * 2 + 1];
    }
    O[r * 2]     = 1.0f / (1.0f + __expf(-s0));
    O[r * 2 + 1] = 1.0f / (1.0f + __expf(-s1));
}

// Convert a [K,N] fp32 weight into swizzled bf16 WMMA B-fragment order:
// tile (kt,nt) holds a 32x16 block, lane-major, 16 contiguous elems per lane.
__global__ void cvt_swz_bf16(const float* __restrict__ src, __bf16* __restrict__ dst,
                             int K, int N) {
    size_t i = (size_t)blockIdx.x * blockDim.x + threadIdx.x;
    if (i >= (size_t)K * N) return;
    int tile = (int)(i >> 9);                      // 512 elems per tile
    int rem  = (int)(i & 511);
    int lane = rem >> 4;
    int e    = rem & 15;
    int tilesN = N >> 4;
    int kt = tile / tilesN, nt = tile % tilesN;
    int k  = (kt << 5) + ((e < 8) ? (((lane >> 4) << 3) + e) : (16 + ((lane >> 4) << 3) + (e - 8)));
    int n  = (nt << 4) + (lane & 15);
    dst[i] = f2bf(src[(size_t)k * N + n]);
}

// ---------------------------------------------------------------------------

extern "C" void kernel_launch(void* const* d_in, const int* in_sizes, int n_in,
                              void* d_out, int out_size, void* d_ws, size_t ws_size,
                              hipStream_t stream) {
    const float* inputs  = (const float*)d_in[0];
    const float* keys    = (const float*)d_in[1];
    const float* values  = (const float*)d_in[2];
    const float* w_first = (const float*)d_in[3];
    const float* b_first = (const float*)d_in[4];
    const float* fc_w    = (const float*)d_in[5];
    const float* fc_b    = (const float*)d_in[6];
    const float* conv_w  = (const float*)d_in[7];
    const float* conv_b  = (const float*)d_in[8];
    const float* att_w1  = (const float*)d_in[9];
    const float* att_b1  = (const float*)d_in[10];
    const float* att_w2  = (const float*)d_in[11];
    const float* att_b2  = (const float*)d_in[12];
    const float* att_w3  = (const float*)d_in[13];
    const float* att_b3  = (const float*)d_in[14];
    const float* att_wo  = (const float*)d_in[15];
    const float* att_bo  = (const float*)d_in[16];
    const float* w_done  = (const float*)d_in[17];
    const float* b_done  = (const float*)d_in[18];
    const float* w_mel   = (const float*)d_in[19];
    const float* b_mel   = (const float*)d_in[20];

    const int B = 32, TQ = 1024, TK = 512;
    const long long sH  = (long long)TQ * 256;     // h batch stride
    const long long sKV = (long long)TK * 256;
    const long long sAW = (long long)TQ * 512;

    char* ws = (char*)d_ws;
    size_t off = 0;
    auto alloc = [&](size_t bytes) -> void* {
        void* p = ws + off;
        off = (off + bytes + 255) & ~(size_t)255;
        return p;
    };

    __bf16* wb_first = (__bf16*)alloc((size_t)320 * 256 * 2);
    __bf16* wb_fc    = (__bf16*)alloc((size_t)3 * 256 * 256 * 2);
    __bf16* wb_conv  = (__bf16*)alloc((size_t)4 * 5 * 256 * 512 * 2);
    __bf16* wb_w1    = (__bf16*)alloc((size_t)4 * 256 * 256 * 2);
    __bf16* wb_w2    = (__bf16*)alloc((size_t)4 * 256 * 256 * 2);
    __bf16* wb_w3    = (__bf16*)alloc((size_t)4 * 256 * 256 * 2);
    __bf16* wb_wo    = (__bf16*)alloc((size_t)4 * 256 * 256 * 2);
    __bf16* wb_mel   = (__bf16*)alloc((size_t)256 * 320 * 2);
    float* buf0  = (float*)alloc((size_t)B * TQ * 256 * 4);
    float* buf1  = (float*)alloc((size_t)B * TQ * 256 * 4);
    float* bufQ  = (float*)alloc((size_t)B * TQ * 256 * 4);
    float* bufK  = (float*)alloc((size_t)B * TK * 256 * 4);
    float* bufV  = (float*)alloc((size_t)B * TK * 256 * 4);   // holds v_out^T [C,Tk]
    float* bufAW = (float*)alloc((size_t)B * TQ * 512 * 4);
    (void)ws_size; (void)n_in; (void)in_sizes; (void)out_size;

    dim3 blk(256);
    auto swz = [&](const float* s, __bf16* d, int K, int N) {
        size_t n = (size_t)K * N;
        cvt_swz_bf16<<<dim3((unsigned)((n + 255) / 256)), blk, 0, stream>>>(s, d, K, N);
    };
    swz(w_first, wb_first, 320, 256);
    for (int i = 0; i < 3; ++i)
        swz(fc_w + (size_t)i * 65536, wb_fc + (size_t)i * 65536, 256, 256);
    for (int i = 0; i < 4; ++i)
        swz(conv_w + (size_t)i * 1280 * 512, wb_conv + (size_t)i * 1280 * 512, 1280, 512);
    for (int i = 0; i < 4; ++i) {
        swz(att_w1 + (size_t)i * 65536, wb_w1 + (size_t)i * 65536, 256, 256);
        swz(att_w2 + (size_t)i * 65536, wb_w2 + (size_t)i * 65536, 256, 256);
        swz(att_w3 + (size_t)i * 65536, wb_w3 + (size_t)i * 65536, 256, 256);
        swz(att_wo + (size_t)i * 65536, wb_wo + (size_t)i * 65536, 256, 256);
    }
    swz(w_mel, wb_mel, 256, 320);

    // h = relu(inputs @ w_first + b_first)          [1024x320 @ 320x256]
    gemm_wmma<1, 0, false, false><<<dim3(32, B), blk, 0, stream>>>(
        inputs, (long long)TQ * 320, 320, wb_first, 0, 16, b_first,
        nullptr, 0, buf0, sH, 256, TQ, 256, 320, 1.0f, 0.0f);

    float* hc = buf0;
    float* hf = buf1;
    for (int i = 0; i < 3; ++i) {                    // FC stack
        gemm_wmma<1, 0, false, false><<<dim3(32, B), blk, 0, stream>>>(
            hc, sH, 256, wb_fc + (size_t)i * 65536, 0, 16, fc_b + i * 256,
            nullptr, 0, hf, sH, 256, TQ, 256, 256, 1.0f, 0.0f);
        float* t = hc; hc = hf; hf = t;
    }

    for (int i = 0; i < 4; ++i) {
        // conv-GLU-residual
        conv_glu_wmma<<<dim3(64, B), blk, 0, stream>>>(
            hc, wb_conv + (size_t)i * 1280 * 512, conv_b + i * 512, hf, TQ, 1 << i);
        { float* t = hc; hc = hf; hf = t; }

        // k_out = (keys + pe(2.0)) @ w1 + b1        [512 rows]
        gemm_wmma<0, 0, true, false><<<dim3(16, B), blk, 0, stream>>>(
            keys, sKV, 256, wb_w1 + (size_t)i * 65536, 0, 16, att_b1 + i * 256,
            nullptr, 0, bufK, sKV, 256, TK, 256, 256, 1.0f, 2.0f);
        // v_out^T = (values @ w3 + b3)^T  -> bufV is [C=256, Tk=512] per batch
        gemm_wmma<0, 0, false, true><<<dim3(16, B), blk, 0, stream>>>(
            values, sKV, 256, wb_w3 + (size_t)i * 65536, 0, 16, att_b3 + i * 256,
            nullptr, 0, bufV, sKV, 512, TK, 256, 256, 1.0f, 0.0f);
        // q_out = (h + pe(1.0)) @ w2 + b2
        gemm_wmma<0, 0, true, false><<<dim3(32, B), blk, 0, stream>>>(
            hc, sH, 256, wb_w2 + (size_t)i * 65536, 0, 16, att_b2 + i * 256,
            nullptr, 0, bufQ, sH, 256, TQ, 256, 256, 1.0f, 1.0f);
        // aw = q_out @ k_out^T                      [1024x512, K=256]
        gemm_wmma<0, 1, false, false><<<dim3(64, B), blk, 0, stream>>>(
            bufQ, sH, 256, bufK, sKV, 256, nullptr,
            nullptr, 0, bufAW, sAW, 512, TQ, 512, 256, 1.0f, 0.0f);
        // softmax rows
        softmax512<<<dim3(4096), blk, 0, stream>>>(bufAW, B * TQ);
        // ctx = (probs @ v_out) * rsqrt(512)  -> overwrite bufQ
        // B(k=tk, n=c) = v_out^T[c*512 + tk]: k-contiguous vector loads
        gemm_wmma<0, 1, false, false><<<dim3(32, B), blk, 0, stream>>>(
            bufAW, sAW, 512, bufV, sKV, 512, nullptr,
            nullptr, 0, bufQ, sH, 256, TQ, 256, 512, 0.04419417382415922f, 0.0f);
        // h = (ctx @ wo + bo + h) * sqrt(0.5)
        gemm_wmma<2, 0, false, false><<<dim3(32, B), blk, 0, stream>>>(
            bufQ, sH, 256, wb_wo + (size_t)i * 65536, 0, 16, att_bo + i * 256,
            hc, sH, hf, sH, 256, TQ, 256, 256, 1.0f, 0.0f);
        { float* t = hc; hc = hf; hf = t; }
    }

    float* out = (float*)d_out;
    float* out_mel  = out;                                   // [32,1024,320]
    float* out_done = out + (size_t)B * TQ * 320;            // [32,1024,2]
    float* out_h    = out_done + (size_t)B * TQ * 2;         // [32,1024,256]

    // mel = h @ w_mel + b_mel                       [N=320]
    gemm_wmma<0, 0, false, false><<<dim3(40, B), blk, 0, stream>>>(
        hc, sH, 256, wb_mel, 0, 20, b_mel,
        nullptr, 0, out_mel, (long long)TQ * 320, 320, TQ, 320, 256, 1.0f, 0.0f);
    // done = sigmoid(h @ w_done + b_done)
    done_head<<<dim3(128), blk, 0, stream>>>(hc, w_done, b_done, out_done, B * TQ);
    // copy final h
    hipMemcpyAsync(out_h, hc, (size_t)B * TQ * 256 * 4, hipMemcpyDeviceToDevice, stream);
}
